// CADRN_1434519077011
// MI455X (gfx1250) — matrix operand order repaired and checked by reading, not compiled, not run on hardware
//
#include <hip/hip_runtime.h>
#include <hip/hip_bf16.h>

typedef __attribute__((ext_vector_type(16))) _Float16 v16h;
typedef __attribute__((ext_vector_type(8)))  float    v8f;
typedef __attribute__((ext_vector_type(2)))  float    f32x2;

// ---------------------------------------------------------------------------
// context_cue: pooled[b,z,y,x] = min_{s=2..5} sum_{|d|inf<=s} mov[b,p+d]*sta[b,C+d]
// One block per (z, b). sta center-crop (11^3) and mov z-slab (11 slices) in LDS.
// ---------------------------------------------------------------------------
__global__ __launch_bounds__(256)
void context_cue_kernel(const float* __restrict__ x, float* __restrict__ pooled) {
    const int z = blockIdx.x;   // 0..31
    const int b = blockIdx.y;   // 0..127
    const float* sta = x;                         // [128][32][32][32]
    const float* mov = x + (size_t)128 * 32768;   // [128][32][32][32]

    __shared__ float sSta[11 * 11 * 11];   // sta[b, 10..20, 10..20, 10..20]
    __shared__ float sMov[11 * 32 * 32];   // mov[b, z-5..z+5, :, :] zero padded

    const int tid = threadIdx.x;
    const size_t base = (size_t)b * 32768;

    for (int i = tid; i < 1331; i += 256) {
        int a = i / 121, r = i - a * 121;
        int bb = r / 11, c = r - bb * 11;
        sSta[i] = sta[base + (size_t)(10 + a) * 1024 + (10 + bb) * 32 + (10 + c)];
    }
    for (int i = tid; i < 11 * 1024; i += 256) {
        int s = i >> 10, p = i & 1023;
        int zz = z + s - 5;
        float v = 0.f;
        if ((unsigned)zz < 32u) v = mov[base + (size_t)zz * 1024 + p];
        sMov[i] = v;
    }
    __syncthreads();

    int py[4], px[4];
#pragma unroll
    for (int j = 0; j < 4; ++j) { int p = tid + 256 * j; py[j] = p >> 5; px[j] = p & 31; }
    float acc2[4] = {0,0,0,0}, acc3[4] = {0,0,0,0}, acc4[4] = {0,0,0,0}, acc5[4] = {0,0,0,0};

    for (int dz = -5; dz <= 5; ++dz) {
        int az = dz < 0 ? -dz : dz;
        const float* movz = &sMov[(dz + 5) * 1024];
        for (int dy = -5; dy <= 5; ++dy) {
            int ay = dy < 0 ? -dy : dy;
            int axy = az > ay ? az : ay;
            const float* staRow = &sSta[(dz + 5) * 121 + (dy + 5) * 11 + 5];
            for (int dx = -5; dx <= 5; ++dx) {
                int ax = dx < 0 ? -dx : dx;
                int mx = axy > ax ? axy : ax;       // uniform over lanes
                float w = staRow[dx];
#pragma unroll
                for (int j = 0; j < 4; ++j) {
                    int yy = py[j] + dy, xx = px[j] + dx;
                    bool ok = ((unsigned)yy < 32u) && ((unsigned)xx < 32u);
                    int idx = ok ? (yy * 32 + xx) : 0;
                    float m = movz[idx];
                    float pr = ok ? (w * m) : 0.f;
                    if (mx <= 2) acc2[j] += pr;
                    if (mx <= 3) acc3[j] += pr;
                    if (mx <= 4) acc4[j] += pr;
                    acc5[j] += pr;
                }
            }
        }
    }
    const size_t ob = base + (size_t)z * 1024;
#pragma unroll
    for (int j = 0; j < 4; ++j)
        pooled[ob + tid + 256 * j] = fminf(fminf(acc2[j], acc3[j]), fminf(acc4[j], acc5[j]));
}

// ---------------------------------------------------------------------------
// concat: h0[b, {0,1,2}, :] = {sta, mov, pooled}
// ---------------------------------------------------------------------------
__global__ void concat_kernel(const float* __restrict__ x, const float* __restrict__ pooled,
                              float* __restrict__ h0) {
    int i = blockIdx.x * blockDim.x + threadIdx.x;
    if (i >= 128 * 32768) return;
    int b = i >> 15, p = i & 32767;
    const float* sta = x;
    const float* mov = x + (size_t)128 * 32768;
    size_t o = (size_t)b * 3 * 32768 + p;
    h0[o]         = sta[i];
    h0[o + 32768] = mov[i];
    h0[o + 65536] = pooled[i];
}

// ---------------------------------------------------------------------------
// Repack conv weights f32 [Cout][Cin*27] -> f16 in this file's contiguous-k WMMA A-fragment order:
//   packed[((ct16*kp + ks)*32 + lane)*16 + e]  with e=2v+ee,
//   K = ks*32 + ((lane>>4)<<4) + e,  M = lane&15.
// ---------------------------------------------------------------------------
__global__ void pack_wgt_kernel(const float* __restrict__ wgt, _Float16* __restrict__ out,
                                int Cout, int Kdim) {
    const int kp = (Kdim + 31) >> 5;
    const int total = (Cout >> 4) * kp * 512;
    int i = blockIdx.x * blockDim.x + threadIdx.x;
    if (i >= total) return;
    int e = i & 15;
    int lane = (i >> 4) & 31;
    int blk = i >> 9;                 // ct16*kp + ks
    int ks = blk % kp, ct = blk / kp;
    int v = e >> 1, ee = e & 1;
    int hh = lane >> 4, r = lane & 15;
    int K = ks * 32 + (hh << 4) + e;
    float val = (K < Kdim) ? wgt[(size_t)(ct * 16 + r) * Kdim + K] : 0.f;
    out[i] = (_Float16)val;
}

// ---------------------------------------------------------------------------
// Implicit-GEMM 3x3x3 conv, pad=1, fused bias+ReLU. One wave per 64(Cout)x16(pos)
// tile: 4 accumulators / 4 A fragments share one im2col B fragment per K-step
// (4x matrix-op density per gather). Cout % 64 == 0 for all layers.
// ---------------------------------------------------------------------------
__global__ __launch_bounds__(32)
void conv3d_wmma_kernel(const float* __restrict__ in, const _Float16* __restrict__ packA,
                        const float* __restrict__ bias, float* __restrict__ out,
                        int Cin, int Cout, int Din, int Do, int stride) {
    const int lane = threadIdx.x;
    const int pt = blockIdx.x, ct = blockIdx.y, b = blockIdx.z;
    const int Do2 = Do * Do, Do3 = Do2 * Do;
    const int Din2 = Din * Din, Din3 = Din2 * Din;
    const int Kdim = Cin * 27;
    const int kp = (Kdim + 31) >> 5;
    const int pos0 = pt * 16, c0 = ct * 64;
    const int r = lane & 15, hh = lane >> 4;

    // this lane's single output position (clamped), input window base coords
    const bool posValid = (pos0 + r) < Do3;
    const int pos = posValid ? (pos0 + r) : 0;
    const int zo = pos / Do2;
    const int rem = pos - zo * Do2;
    const int yo = rem / Do, xo = rem - yo * Do;
    const int zbs = zo * stride - 1, ybs = yo * stride - 1, xbs = xo * stride - 1;
    const float* inB = in + (size_t)b * Cin * Din3;

    const v16h* pa = (const v16h*)packA + lane;

    v8f acc[4] = {{}, {}, {}, {}};
    for (int ks = 0; ks < kp; ++ks) {
        // ---- shared im2col B fragment (K = ks*32 + 16*half + i), branchless
        v16h bf;
        const int kbase = ks * 32 + (hh << 4);
#pragma unroll
        for (int i = 0; i < 16; ++i) {
            int k = kbase + i;
            int cin = k / 27;  int t  = k - cin * 27;
            int dz  = t / 9;   int rr = t - dz * 9;
            int dy  = rr / 3;  int dx = rr - dy * 3;
            int zi = zbs + dz, yi = ybs + dy, xi = xbs + dx;
            bool ok = posValid && (k < Kdim) &&
                      (unsigned)zi < (unsigned)Din &&
                      (unsigned)yi < (unsigned)Din &&
                      (unsigned)xi < (unsigned)Din;
            long idx = (long)cin * Din3 + (long)zi * Din2 + (long)(yi * Din + xi);
            idx = ok ? idx : 0;
            float v = inB[idx];
            bf[i] = (_Float16)(ok ? v : 0.f);
        }
        __builtin_prefetch((const void*)(pa + ((size_t)(ct * 4) * kp + ks + 1) * 32), 0, 0);
        // ---- 4 register-direct A fragments, 4 WMMAs against shared bf
#pragma unroll
        for (int j = 0; j < 4; ++j) {
            v16h af = pa[((size_t)(ct * 4 + j) * kp + ks) * 32];
            acc[j] = __builtin_amdgcn_wmma_f32_16x16x32_f16(false, af, false, bf,
                                                            (short)0, acc[j], false, false);
        }
    }
    if (posValid) {
#pragma unroll
        for (int j = 0; j < 4; ++j) {
#pragma unroll
            for (int v = 0; v < 8; ++v) {
                int co = c0 + j * 16 + v + 8 * hh;
                float val = acc[j][v] + bias[co];
                out[((size_t)b * Cout + co) * Do3 + pos] = fmaxf(val, 0.f);
            }
        }
    }
}

// ---------------------------------------------------------------------------
// maxpool 3x3x3 stride 2 pad 1: [128,128,8^3] -> [128,128,4^3]
// ---------------------------------------------------------------------------
__global__ void maxpool3_kernel(const float* __restrict__ in, float* __restrict__ out) {
    int i = blockIdx.x * blockDim.x + threadIdx.x;
    if (i >= 128 * 128 * 64) return;
    int x = i & 3, y = (i >> 2) & 3, z = (i >> 4) & 3, c = (i >> 6) & 127, b = i >> 13;
    const float* p = in + ((size_t)(b * 128 + c) << 9);
    float m = -3.402823466e38f;
#pragma unroll
    for (int kz = 0; kz < 3; ++kz) {
#pragma unroll
        for (int ky = 0; ky < 3; ++ky) {
#pragma unroll
            for (int kx = 0; kx < 3; ++kx) {
                int zz = 2 * z - 1 + kz, yy = 2 * y - 1 + ky, xx = 2 * x - 1 + kx;
                bool ok = (unsigned)zz < 8u && (unsigned)yy < 8u && (unsigned)xx < 8u;
                int idx = ok ? ((zz << 6) + (yy << 3) + xx) : 0;
                float v = p[idx];
                m = fmaxf(m, ok ? v : -3.402823466e38f);
            }
        }
    }
    out[i] = m;
}

// ---------------------------------------------------------------------------
// FC GEMM: out[m,n] = sum_k X[m,k]*W[n,k] + bias[n], M = 128, K % 32 == 0.
// One wave per 64(M)x16(N) tile: 4 accumulators / 4 A row-sets share one
// B (weight) fragment per K-step. Register-direct float2 fragment loads.
// ---------------------------------------------------------------------------
__global__ __launch_bounds__(32)
void gemm_bias_wmma_kernel(const float* __restrict__ X, const float* __restrict__ W,
                           const float* __restrict__ bias, float* __restrict__ out,
                           int N, int K) {
    const int lane = threadIdx.x;
    const int n0 = blockIdx.x * 16, m0 = blockIdx.y * 64;
    const int r = lane & 15, hh = lane >> 4;

    const bool nValid = (n0 + r) < N;
    const int n = nValid ? (n0 + r) : 0;
    const f32x2* Wr = (const f32x2*)(W + (size_t)n * K);
    const f32x2* Xr[4];
#pragma unroll
    for (int j = 0; j < 4; ++j)
        Xr[j] = (const f32x2*)(X + (size_t)(m0 + j * 16 + r) * K);

    v8f acc[4] = {{}, {}, {}, {}};
    for (int k0 = 0; k0 < K; k0 += 32) {
        // shared weight fragment
        v16h bf;
#pragma unroll
        for (int v = 0; v < 8; ++v) {
            int kb = k0 + ((v & 4) << 2) + ((v & 3) << 1) + (hh << 3);
            f32x2 wv = Wr[kb >> 1];
            bf[2 * v]     = (_Float16)(nValid ? wv.x : 0.f);
            bf[2 * v + 1] = (_Float16)(nValid ? wv.y : 0.f);
        }
#pragma unroll
        for (int j = 0; j < 4; ++j) {
            v16h af;
#pragma unroll
            for (int v = 0; v < 8; ++v) {
                int ka = k0 + ((v & 4) << 2) + ((v & 3) << 1) + (hh << 3);
                f32x2 xv = Xr[j][ka >> 1];
                af[2 * v]     = (_Float16)xv.x;
                af[2 * v + 1] = (_Float16)xv.y;
            }
            acc[j] = __builtin_amdgcn_wmma_f32_16x16x32_f16(false, af, false, bf,
                                                            (short)0, acc[j], false, false);
        }
    }
    if (nValid) {
        float bv = bias[n];
#pragma unroll
        for (int j = 0; j < 4; ++j)
#pragma unroll
            for (int v = 0; v < 8; ++v)
                out[(size_t)(m0 + j * 16 + v + 8 * hh) * N + n] = acc[j][v] + bv;
    }
}

// ---------------------------------------------------------------------------
// Batch-norm over batch dim (M=128) + ReLU/tanh. One thread per feature.
// ---------------------------------------------------------------------------
__global__ void bn_act_kernel(const float* __restrict__ y, const float* __restrict__ g,
                              const float* __restrict__ be, float* __restrict__ out,
                              int N, int tanhAct) {
    int f = blockIdx.x * blockDim.x + threadIdx.x;
    if (f >= N) return;
    float s = 0.f, s2 = 0.f;
    for (int i = 0; i < 128; ++i) { float v = y[(size_t)i * N + f]; s += v; s2 += v * v; }
    float m = s * (1.f / 128.f);
    float var = s2 * (1.f / 128.f) - m * m;
    float inv = rsqrtf(var + 1e-5f);
    float gg = g[f], bb = be[f];
    for (int i = 0; i < 128; ++i) {
        float v = (y[(size_t)i * N + f] - m) * inv * gg + bb;
        out[(size_t)i * N + f] = tanhAct ? tanhf(v) : fmaxf(v, 0.f);
    }
}

// ---------------------------------------------------------------------------
extern "C" void kernel_launch(void* const* d_in, const int* in_sizes, int n_in,
                              void* d_out, int out_size, void* d_ws, size_t ws_size,
                              hipStream_t stream) {
    (void)in_sizes; (void)n_in; (void)out_size; (void)ws_size;
    const float* x = (const float*)d_in[0];

    float* W0     = (float*)d_ws;
    float* pooled = W0;                  // 4,194,304 f32
    float* h0     = W0 + 4194304;        // 12,582,912 f32
    float* bufA   = W0 + 16777216;       // 33,554,432 f32
    float* bufB   = W0 + 50331648;       // 33,554,432 f32
    _Float16* packW = (_Float16*)(W0 + 83886080);  // ~14.1M f16 packed conv weights

    // --- pack all conv weights into fragment order (offsets in f16 elements)
    struct { int wi, Cin, Cout; } layers[8] = {
        {1, 3, 64}, {3, 64, 64}, {5, 64, 128}, {7, 128, 128},
        {9, 128, 256}, {11, 256, 256}, {13, 256, 512}, {15, 512, 512}};
    size_t poff[8];
    size_t acc = 0;
    for (int l = 0; l < 8; ++l) {
        poff[l] = acc;
        int Kdim = layers[l].Cin * 27;
        int kp = (Kdim + 31) >> 5;
        size_t total = (size_t)(layers[l].Cout >> 4) * kp * 512;
        pack_wgt_kernel<<<((int)total + 255) / 256, 256, 0, stream>>>(
            (const float*)d_in[layers[l].wi], packW + poff[l], layers[l].Cout, Kdim);
        acc += total;
    }

    context_cue_kernel<<<dim3(32, 128), 256, 0, stream>>>(x, pooled);
    concat_kernel<<<(128 * 32768 + 255) / 256, 256, 0, stream>>>(x, pooled, h0);

    auto conv = [&](const float* in, int l, float* out,
                    int Din, int Do, int stride) {
        int Cin = layers[l].Cin, Cout = layers[l].Cout;
        dim3 grid((Do * Do * Do + 15) / 16, Cout / 64, 128);
        conv3d_wmma_kernel<<<grid, 32, 0, stream>>>(
            in, packW + poff[l], (const float*)d_in[layers[l].wi + 1], out,
            Cin, Cout, Din, Do, stride);
    };

    conv(h0,   0, bufA, 32, 16, 2);   // c1a -> t1
    conv(bufA, 1, bufB, 16, 16, 1);   // c1b -> h1
    conv(bufB, 2, bufA, 16, 8,  2);   // c2a -> t2
    conv(bufA, 3, W0,   8,  8,  1);   // c2b -> h2 (pooled/h0 dead)
    maxpool3_kernel<<<(128 * 128 * 64 + 255) / 256, 256, 0, stream>>>(W0, bufB); // p
    conv(bufB, 4, bufA, 4,  2,  2);   // c3a -> t3
    conv(bufA, 5, W0,   2,  2,  1);   // c3b -> h3
    conv(W0,   6, bufB, 2,  1,  2);   // c4a -> t4
    conv(bufB, 7, bufA, 1,  1,  1);   // c4b -> h4 = [128,512]

    auto fc = [&](const float* in, int wi, float* y, float* dst, int N, int K, int tanhAct) {
        dim3 grid((N + 15) / 16, 2);  // 2 = M(128)/64
        gemm_bias_wmma_kernel<<<grid, 32, 0, stream>>>(
            in, (const float*)d_in[wi], (const float*)d_in[wi + 1], y, N, K);
        bn_act_kernel<<<(N + 255) / 256, 256, 0, stream>>>(
            y, (const float*)d_in[wi + 2], (const float*)d_in[wi + 3], dst, N, tanhAct);
    };

    fc(bufA, 17, W0,   bufB, 4096, 512,  0);          // fc1
    fc(bufB, 21, bufA, W0,   1024, 4096, 0);          // fc2
    fc(W0,   25, bufB, bufA, 128,  1024, 0);          // fc3
    fc(bufA, 29, W0, (float*)d_out, 3, 128, 1);       // fc4 + tanh -> d_out [128,3]
}
